// BeamformingTransformer_76699525972172
// MI455X (gfx1250) — compile-verified, matrix operand order checked by reading, not run on hardware
//
#include <hip/hip_runtime.h>
#include <hip/hip_bf16.h>
#include <math.h>

#define USE_ASYNC_LDS 1   // inline-asm global_load_async_to_lds_b128 staging

// ---------------------------------------------------------------------------
// Model constants (from the reference)
// ---------------------------------------------------------------------------
#define D_MODEL 512
#define NHEAD   8
#define HDIM    64
#define NLAYER  6
#define KUSR    16
#define NTX     64
#define MSLQ    138
#define RELW    (2*MSLQ - 1)
#define BATCH   64
#define SCALE_ATT 0.125f   // 1/sqrt(64)

// ---------------------------------------------------------------------------
// WMMA types (gfx1250, wave32)
// ---------------------------------------------------------------------------
typedef __attribute__((ext_vector_type(16))) __bf16 v16bf;
typedef __attribute__((ext_vector_type(8)))  float  v8f;

#define BM 128
#define BN 128
#define BK 32
#define GEMM_THREADS 256

#define EP_GELU 1
#define EP_RES  2
#define EP_ACC  4

__device__ __forceinline__ float gelu_f(float x) {
    return 0.5f * x * (1.0f + erff(x * 0.70710678118654752440f));
}

// pack two fp32 into a dword of two bf16
__device__ __forceinline__ unsigned pack_bf16(float a, float b) {
    union { __bf16 h[2]; unsigned u; } p;
    p.h[0] = (__bf16)a;
    p.h[1] = (__bf16)b;
    return p.u;
}

// ---------------------------------------------------------------------------
// Fragment swizzle (CDNA5 ISA 7.12.2), 16-bit operands, 16x32 tiles:
//   lanes 0-15 hold K {0..7, 16..23}, lanes 16-31 hold K {8..15, 24..31},
//   two K packed per dword.  K-pair p (k = 2p): dword = ((p&8)>>1)+(p&3),
//   laneHalf = (p&4)?16:0.  K-octet o (k = 8o..8o+7) maps to one contiguous
//   16-byte region: laneHalf = (o&1)?16:0, dword base = (o&2)?4:0.
// Slab layout (dwords): tile(8) x lane(32) x dword(8) = 2048 dwords = 8 KB.
// ---------------------------------------------------------------------------

// ---- fast A staging: full tile, K multiple of BK ----
__device__ __forceinline__ void stage_fast_A(unsigned* __restrict__ lsA,
                                             const float* __restrict__ Ap,
                                             int m0, int k0, int lda, int tid)
{
    #pragma unroll
    for (int i = tid; i < BM * 4; i += GEMM_THREADS) {
        int m = i >> 2, o = i & 3;
        const float* src = Ap + (long long)(m0 + m) * lda + k0 + o * 8;
        float4 f0 = *(const float4*)src;
        float4 f1 = *(const float4*)(src + 4);
        uint4 p;
        p.x = pack_bf16(f0.x, f0.y); p.y = pack_bf16(f0.z, f0.w);
        p.z = pack_bf16(f1.x, f1.y); p.w = pack_bf16(f1.z, f1.w);
        int lane = (m & 15) + ((o & 1) ? 16 : 0);
        int dwb  = (o & 2) ? 4 : 0;
        *(uint4*)&lsA[((m >> 4) << 8) + (lane << 3) + dwb] = p;
    }
}

// ---- fast B staging from fp32 (TRANSB static) ----
template <bool TRANSB>
__device__ __forceinline__ void stage_fast_B(unsigned* __restrict__ lsB,
                                             const float* __restrict__ Bp,
                                             int n0, int k0, int ldb, int tid)
{
    #pragma unroll
    for (int i = tid; i < BN * 4; i += GEMM_THREADS) {
        int n = i & 127, o = i >> 7;
        uint4 p;
        if (TRANSB) {
            const float* src = Bp + (long long)(n0 + n) * ldb + k0 + o * 8;
            float4 f0 = *(const float4*)src;
            float4 f1 = *(const float4*)(src + 4);
            p.x = pack_bf16(f0.x, f0.y); p.y = pack_bf16(f0.z, f0.w);
            p.z = pack_bf16(f1.x, f1.y); p.w = pack_bf16(f1.z, f1.w);
        } else {
            const float* src = Bp + (long long)(k0 + o * 8) * ldb + (n0 + n);
            float v[8];
            #pragma unroll
            for (int e = 0; e < 8; ++e) v[e] = src[(long long)e * ldb];
            p.x = pack_bf16(v[0], v[1]); p.y = pack_bf16(v[2], v[3]);
            p.z = pack_bf16(v[4], v[5]); p.w = pack_bf16(v[6], v[7]);
        }
        int lane = (n & 15) + ((o & 1) ? 16 : 0);
        int dwb  = (o & 2) ? 4 : 0;
        *(uint4*)&lsB[((n >> 4) << 8) + (lane << 3) + dwb] = p;
    }
}

// ---- B staging from pre-swizzled bf16 weights: contiguous 8 KB copy ----
__device__ __forceinline__ void stage_psb_B(unsigned* __restrict__ lsB,
                                            const unsigned* __restrict__ chunk,
                                            int tid)
{
#if USE_ASYNC_LDS
    // ASYNCcnt-tracked DMA, bypasses VGPRs (ISA 15.18.3 op 98)
    unsigned ldsBase = (unsigned)(size_t)lsB;   // LDS aperture: low 32 bits
    #pragma unroll
    for (int i = tid; i < 512; i += GEMM_THREADS) {
        asm volatile("global_load_async_to_lds_b128 %0, %1, off"
                     :: "v"(ldsBase + i * 16), "v"(chunk + i * 4) : "memory");
    }
#else
    #pragma unroll
    for (int i = tid; i < 512; i += GEMM_THREADS)
        ((uint4*)lsB)[i] = ((const uint4*)chunk)[i];
#endif
}

__device__ __forceinline__ void async_fence()
{
#if USE_ASYNC_LDS
    asm volatile("s_wait_asynccnt 0x0" ::: "memory");
#endif
}

// ---- slow staging: K-pair per thread, flat predicated loads ----
__device__ __forceinline__ void stage_slow(unsigned* __restrict__ lsA,
                                           unsigned* __restrict__ lsB,
                                           const float* __restrict__ Ap,
                                           const float* __restrict__ Bp,
                                           int m0, int n0, int k0,
                                           int M, int N, int K,
                                           int lda, int ldb, int transB, int tid)
{
    #pragma unroll
    for (int i = tid; i < (BM * BK) / 2; i += GEMM_THREADS) {
        int m = i >> 4, kp = i & 15;
        int gm = m0 + m, gk = k0 + (kp << 1);
        bool rOk = gm < M;
        const float* src = Ap + (long long)gm * lda + gk;
        float v0 = (rOk && gk < K)     ? src[0] : 0.f;
        float v1 = (rOk && gk + 1 < K) ? src[1] : 0.f;
        int dw   = ((kp & 8) >> 1) + (kp & 3);
        int lane = (m & 15) + ((kp & 4) ? 16 : 0);
        lsA[((m >> 4) << 8) + (lane << 3) + dw] = pack_bf16(v0, v1);
    }
    #pragma unroll
    for (int i = tid; i < (BK * BN) / 2; i += GEMM_THREADS) {
        int n = i & 127, kp = i >> 7;
        int gk = k0 + (kp << 1), gn = n0 + n;
        bool cOk = gn < N;
        float v0, v1;
        if (transB) {
            const float* src = Bp + (long long)gn * ldb + gk;
            v0 = (cOk && gk < K)     ? src[0] : 0.f;
            v1 = (cOk && gk + 1 < K) ? src[1] : 0.f;
        } else {
            v0 = (cOk && gk < K)     ? Bp[(long long)gk * ldb + gn]       : 0.f;
            v1 = (cOk && gk + 1 < K) ? Bp[(long long)(gk + 1) * ldb + gn] : 0.f;
        }
        int dw   = ((kp & 8) >> 1) + (kp & 3);
        int lane = (n & 15) + ((kp & 4) ? 16 : 0);
        lsB[((n >> 4) << 8) + (lane << 3) + dw] = pack_bf16(v0, v1);
    }
}

// ---- one K-step of WMMAs from a staged slab ----
__device__ __forceinline__ void compute_step(const unsigned* __restrict__ lsA,
                                             const unsigned* __restrict__ lsB,
                                             int lane, int wm, int wn,
                                             v8f acc[2][4])
{
    const v16bf* pa = reinterpret_cast<const v16bf*>(lsA);
    const v16bf* pb = reinterpret_cast<const v16bf*>(lsB);
    v16bf af[2], bfr[4];
    #pragma unroll
    for (int i = 0; i < 2; ++i) af[i]  = pa[((wm * 2 + i) << 5) + lane];
    #pragma unroll
    for (int j = 0; j < 4; ++j) bfr[j] = pb[((wn * 4 + j) << 5) + lane];
    #pragma unroll
    for (int i = 0; i < 2; ++i)
        #pragma unroll
        for (int j = 0; j < 4; ++j)
            acc[i][j] = __builtin_amdgcn_wmma_f32_16x16x32_bf16(
                false, af[i], false, bfr[j], (short)0, acc[i][j], false, false);
}

template <bool TRANSB>
__device__ __forceinline__ void gemm_loop_fast(unsigned (*lsA)[2048],
                                               unsigned (*lsB)[2048],
                                               const float* Ap, const float* Bp,
                                               int m0, int n0, int K,
                                               int lda, int ldb,
                                               int tid, int lane, int wm, int wn,
                                               v8f acc[2][4])
{
    const int nK = K / BK;
    stage_fast_A(lsA[0], Ap, m0, 0, lda, tid);
    stage_fast_B<TRANSB>(lsB[0], Bp, n0, 0, ldb, tid);
    __syncthreads();
    for (int tk = 0; tk < nK; ++tk) {
        const int cur = tk & 1;
        if (tk + 1 < nK) {
            stage_fast_A(lsA[cur ^ 1], Ap, m0, (tk + 1) * BK, lda, tid);
            stage_fast_B<TRANSB>(lsB[cur ^ 1], Bp, n0, (tk + 1) * BK, ldb, tid);
        }
        if (tk + 2 < nK && (tid & 1) == 0)   // pull slab tk+2 toward L2
            __builtin_prefetch(Ap + (long long)(m0 + (tid >> 1)) * lda
                                  + (tk + 2) * BK, 0, 1);
        compute_step(lsA[cur], lsB[cur], lane, wm, wn, acc);
        __syncthreads();
    }
}

// B from pre-swizzled bf16 weights; chunk tk covers cols n0..n0+127
__device__ __forceinline__ void gemm_loop_psb(unsigned (*lsA)[2048],
                                              unsigned (*lsB)[2048],
                                              const float* Ap,
                                              const unsigned* swzB,
                                              int m0, int n0, int K, int lda,
                                              int nBlocks,
                                              int tid, int lane, int wm, int wn,
                                              v8f acc[2][4])
{
    const int nK = K / BK;
    const int nb = n0 >> 7;
    stage_fast_A(lsA[0], Ap, m0, 0, lda, tid);
    stage_psb_B(lsB[0], swzB + (long long)nb * 2048, tid);
    async_fence();
    __syncthreads();
    for (int tk = 0; tk < nK; ++tk) {
        const int cur = tk & 1;
        if (tk + 1 < nK) {
            stage_fast_A(lsA[cur ^ 1], Ap, m0, (tk + 1) * BK, lda, tid);
            stage_psb_B(lsB[cur ^ 1],
                        swzB + ((long long)(tk + 1) * nBlocks + nb) * 2048, tid);
        }
        if (tk + 2 < nK && (tid & 1) == 0)
            __builtin_prefetch(Ap + (long long)(m0 + (tid >> 1)) * lda
                                  + (tk + 2) * BK, 0, 1);
        compute_step(lsA[cur], lsB[cur], lane, wm, wn, acc);
        async_fence();          // next-slab DMA complete before buffer swap
        __syncthreads();
    }
}

__device__ __forceinline__ void gemm_loop_slow(unsigned (*lsA)[2048],
                                               unsigned (*lsB)[2048],
                                               const float* Ap, const float* Bp,
                                               int m0, int n0, int M, int N, int K,
                                               int lda, int ldb, int transB,
                                               int tid, int lane, int wm, int wn,
                                               v8f acc[2][4])
{
    const int nK = (K + BK - 1) / BK;
    stage_slow(lsA[0], lsB[0], Ap, Bp, m0, n0, 0, M, N, K, lda, ldb, transB, tid);
    __syncthreads();
    for (int tk = 0; tk < nK; ++tk) {
        const int cur = tk & 1;
        if (tk + 1 < nK)
            stage_slow(lsA[cur ^ 1], lsB[cur ^ 1], Ap, Bp, m0, n0, (tk + 1) * BK,
                       M, N, K, lda, ldb, transB, tid);
        compute_step(lsA[cur], lsB[cur], lane, wm, wn, acc);
        __syncthreads();
    }
}

// ---------------------------------------------------------------------------
// Batched GEMM: C = A x B (+bias +residual +accum, optional GELU)
//   fp32 (or pre-swizzled bf16) operands, bf16 LDS double-buffered, f32 acc.
//   8 waves per 128x128 tile, each wave 32x64 = 8 x v_wmma_f32_16x16x32_bf16.
// ---------------------------------------------------------------------------
__global__ __launch_bounds__(GEMM_THREADS)
void wmma_gemm_kernel(float* __restrict__ C, const float* __restrict__ A,
                      const float* __restrict__ B, const float* __restrict__ bias,
                      const float* __restrict__ Res,
                      int M, int N, int K, int lda, int ldb, int ldc,
                      int nh,
                      long long aZ, long long aH, long long bZ, long long bH,
                      long long cZ, long long cH, long long rZ,
                      int transB, int psb, int flags)
{
    __shared__ alignas(32) unsigned lsA32[2][2048];   // 2 x 8 KB
    __shared__ alignas(32) unsigned lsB32[2][2048];   // 2 x 8 KB

    const int tid  = threadIdx.x;
    const int lane = tid & 31;
    const int wave = tid >> 5;       // 8 waves
    const int wm   = wave >> 1;      // 0..3 : 32-row strip
    const int wn   = wave & 1;       // 0..1 : 64-col strip

    const int z  = blockIdx.z;
    const int bb = z / nh, hh = z % nh;
    const float* Ap = A + (long long)bb * aZ + (long long)hh * aH;
    const float* Bp = B + (long long)bb * bZ + (long long)hh * bH;
    float*       Cp = C + (long long)bb * cZ + (long long)hh * cH;
    const float* Rp = Res ? (Res + (long long)bb * rZ) : nullptr;

    const int m0 = blockIdx.y * BM;
    const int n0 = blockIdx.x * BN;

    v8f acc[2][4] = {};

    const bool full = (m0 + BM <= M) && (n0 + BN <= N) && ((K & (BK - 1)) == 0);
    if (psb) {        // pre-swizzled bf16 weights (always full tiles)
        gemm_loop_psb(lsA32, lsB32, Ap, (const unsigned*)Bp, m0, n0, K, lda,
                      N >> 7, tid, lane, wm, wn, acc);
    } else if (full) {
        if (transB)
            gemm_loop_fast<true >(lsA32, lsB32, Ap, Bp, m0, n0, K, lda, ldb,
                                  tid, lane, wm, wn, acc);
        else
            gemm_loop_fast<false>(lsA32, lsB32, Ap, Bp, m0, n0, K, lda, ldb,
                                  tid, lane, wm, wn, acc);
    } else {
        gemm_loop_slow(lsA32, lsB32, Ap, Bp, m0, n0, M, N, K, lda, ldb, transB,
                       tid, lane, wm, wn, acc);
    }

    // ---- epilogue (C/D layout: lane<16 rows r, lane>=16 rows r+8) ----
    const int rofs = (lane >> 4) * 8;
    const int c0   = lane & 15;
    #pragma unroll
    for (int i = 0; i < 2; ++i) {
        #pragma unroll
        for (int j = 0; j < 4; ++j) {
            int col = n0 + ((wn * 4 + j) << 4) + c0;
            int rowBase = m0 + ((wm * 2 + i) << 4) + rofs;
            if (col < N) {
                const float bv = bias ? bias[col] : 0.0f;
                #pragma unroll
                for (int r = 0; r < 8; ++r) {
                    int row = rowBase + r;
                    if (row < M) {
                        float v = acc[i][j][r] + bv;
                        if (flags & EP_RES)  v += Rp[(long long)row * ldc + col];
                        if (flags & EP_ACC)  v += Cp[(long long)row * ldc + col];
                        if (flags & EP_GELU) v  = gelu_f(v);
                        Cp[(long long)row * ldc + col] = v;
                    }
                }
            }
        }
    }
}

// ---------------------------------------------------------------------------
// Weight pre-swizzle: W (KxN fp32, row-major) -> bf16 fragment-ordered chunks.
// One workgroup per (kSlab, nBlock) = 2048 dwords = 8 KB chunk.
// ---------------------------------------------------------------------------
__global__ __launch_bounds__(256)
void swizzle_weight_kernel(unsigned* __restrict__ dst,
                           const float* __restrict__ W, int K, int N)
{
    const int nBlocks = N >> 7;
    const int ks = blockIdx.x / nBlocks;
    const int nb = blockIdx.x % nBlocks;
    unsigned* out = dst + (long long)blockIdx.x * 2048;
    for (int d = threadIdx.x; d < 2048; d += 256) {
        int ntile = d >> 8, rem = d & 255;
        int lane = rem >> 3, dw = rem & 7;
        int half = lane >> 4;
        int n = nb * 128 + ntile * 16 + (lane & 15);
        int p = (dw & 3) | ((dw & 4) << 1) | (half << 2);   // K-pair index
        int k = ks * 32 + (p << 1);
        float v0 = W[(long long)k * N + n];
        float v1 = W[(long long)(k + 1) * N + n];
        out[d] = pack_bf16(v0, v1);
    }
}

// ---------------------------------------------------------------------------
// Row LayerNorm (optionally fused exact GELU), one workgroup per row
// ---------------------------------------------------------------------------
__global__ __launch_bounds__(256)
void layernorm_kernel(float* __restrict__ out, const float* __restrict__ in,
                      const float* __restrict__ g, const float* __restrict__ b,
                      int F, int doGelu)
{
    __shared__ float s1[256], s2[256];
    const long long row = blockIdx.x;
    const float* x = in + row * F;
    float*       o = out + row * F;
    float sum = 0.f, sq = 0.f;
    for (int i = threadIdx.x; i < F; i += 256) { float v = x[i]; sum += v; sq += v * v; }
    s1[threadIdx.x] = sum; s2[threadIdx.x] = sq;
    __syncthreads();
    for (int off = 128; off > 0; off >>= 1) {
        if (threadIdx.x < off) { s1[threadIdx.x] += s1[threadIdx.x + off];
                                 s2[threadIdx.x] += s2[threadIdx.x + off]; }
        __syncthreads();
    }
    float mean = s1[0] / F;
    float var  = s2[0] / F - mean * mean;
    float rstd = rsqrtf(var + 1e-5f);
    for (int i = threadIdx.x; i < F; i += 256) {
        float v = (x[i] - mean) * rstd * g[i] + b[i];
        if (doGelu) v = gelu_f(v);
        o[i] = v;
    }
}

// ---------------------------------------------------------------------------
// scale + bias + softmax over score rows.  mode 0: cross_bias[h,q,k]
//                                          mode 1: rel_bias[h, clip(k-q+MSL-1)]
// ---------------------------------------------------------------------------
__global__ __launch_bounds__(32)
void attn_softmax_kernel(float* __restrict__ sc,
                         const float* __restrict__ cross_bias,
                         const float* __restrict__ rel_bias,
                         int Lq, int Lk, int mode)
{
    const int q = blockIdx.x;
    const int z = blockIdx.y;          // b*NHEAD + h
    const int h = z % NHEAD;
    const int lane = threadIdx.x;
    float* row = sc + ((long long)z * Lq + q) * Lk;

    float vals[4];
    float mx = -1e30f;
    int nIt = (Lk + 31) >> 5;
    for (int it = 0; it < nIt; ++it) {
        int kk = lane + (it << 5);
        float v = -1e30f;
        if (kk < Lk) {
            float bsv;
            if (mode == 0) {
                bsv = cross_bias[((long long)h * MSLQ + q) * MSLQ + kk];
            } else {
                int idx = kk - q + (MSLQ - 1);
                idx = idx < 0 ? 0 : (idx > RELW - 1 ? RELW - 1 : idx);
                bsv = rel_bias[h * RELW + idx];
            }
            v = row[kk] * SCALE_ATT + bsv;
        }
        vals[it] = v;
        mx = fmaxf(mx, v);
    }
    for (int off = 16; off > 0; off >>= 1) mx = fmaxf(mx, __shfl_xor(mx, off, 32));
    float sm = 0.f;
    for (int it = 0; it < nIt; ++it) {
        int kk = lane + (it << 5);
        float e = (kk < Lk) ? __expf(vals[it] - mx) : 0.f;
        vals[it] = e; sm += e;
    }
    for (int off = 16; off > 0; off >>= 1) sm += __shfl_xor(sm, off, 32);
    float inv = 1.f / sm;
    for (int it = 0; it < nIt; ++it) {
        int kk = lane + (it << 5);
        if (kk < Lk) row[kk] = vals[it] * inv;
    }
}

// ---------------------------------------------------------------------------
// Build stream inputs:
//  mode 0 (antenna): out[b, t(2N), f(K)] = {re|im}[b, f, t%N]   (transposed)
//  mode 1 (user):    out[b, t(2K), f(N)] = {re|im}[b, t%K, f]
// ---------------------------------------------------------------------------
__global__ void gather_stream_kernel(float* __restrict__ out,
                                     const float* __restrict__ re,
                                     const float* __restrict__ im,
                                     int mode, int total)
{
    int idx = blockIdx.x * blockDim.x + threadIdx.x;
    if (idx >= total) return;
    if (mode == 0) {
        int f = idx % KUSR;
        int t = (idx / KUSR) % (2 * NTX);
        int b = idx / (KUSR * 2 * NTX);
        const float* src = (t < NTX) ? re : im;
        int tt = (t < NTX) ? t : t - NTX;
        out[idx] = src[((long long)b * KUSR + f) * NTX + tt];
    } else {
        int f = idx % NTX;
        int t = (idx / NTX) % (2 * KUSR);
        int b = idx / (NTX * 2 * KUSR);
        const float* src = (t < KUSR) ? re : im;
        int tt = (t < KUSR) ? t : t - KUSR;
        out[idx] = src[((long long)b * KUSR + tt) * NTX + f];
    }
}

// ---------------------------------------------------------------------------
// Frobenius-normalize vec (B, 2*K*N) and emit (B, N, K, 2)
// ---------------------------------------------------------------------------
__global__ __launch_bounds__(256)
void finalize_kernel(float* __restrict__ out, const float* __restrict__ vec)
{
    __shared__ float s1[256];
    const int b = blockIdx.x;
    const float* v = vec + (long long)b * 2 * KUSR * NTX;
    float sq = 0.f;
    for (int i = threadIdx.x; i < 2 * KUSR * NTX; i += 256) { float x = v[i]; sq += x * x; }
    s1[threadIdx.x] = sq; __syncthreads();
    for (int off = 128; off > 0; off >>= 1) {
        if (threadIdx.x < off) s1[threadIdx.x] += s1[threadIdx.x + off];
        __syncthreads();
    }
    float inv = 1.f / sqrtf(s1[0]);
    const int KN = KUSR * NTX;
    for (int i = threadIdx.x; i < KN; i += 256) {
        int k = i / NTX, n = i % NTX;
        long long o = (((long long)b * NTX + n) * KUSR + k) * 2;
        out[o]     = v[i]      * inv;   // real
        out[o + 1] = v[KN + i] * inv;   // imag
    }
}

// ---------------------------------------------------------------------------
// Host-side orchestration
// ---------------------------------------------------------------------------
struct Lin { const float* W; const float* b; };
struct LnP { const float* g; const float* b; };
struct Blk {
    Lin q, k, v, proj;
    const float* cross_bias;
    const float* rel_bias;
    LnP ln_self, ln1_q, ln1_k, ln1_v, ln2;
    Lin mlp0, mlp1, mlp2;
};
struct Scratch { float *ln, *q, *k, *v, *s, *y, *o1, *o2, *h1, *h2; };
struct SwzW { unsigned *q, *k, *v, *proj, *m0, *m1, *m2; };

static inline void gemm(hipStream_t st, float* C, const float* A, const float* B,
                        const float* bias, const float* Res,
                        int M, int N, int K, int lda, int ldb, int ldc,
                        int Z, int nh,
                        long long aZ, long long aH, long long bZ, long long bH,
                        long long cZ, long long cH, long long rZ,
                        int transB, int flags, int psb = 0)
{
    dim3 g((N + BN - 1) / BN, (M + BM - 1) / BM, Z);
    wmma_gemm_kernel<<<g, GEMM_THREADS, 0, st>>>(C, A, B, bias, Res,
        M, N, K, lda, ldb, ldc, nh, aZ, aH, bZ, bH, cZ, cH, rZ, transB, psb, flags);
}

static inline void gemm_tok(hipStream_t st, float* C, const float* A, const float* W,
                            const float* bias, const float* Res,
                            int M, int N, int K, int flags)
{
    gemm(st, C, A, W, bias, Res, M, N, K, K, N, N,
         1, 1, 0, 0, 0, 0, 0, 0, 0, 0, flags, 0);
}

// token linear with pre-swizzled bf16 weights (M,N mult of 128, K mult of 32)
static inline void gemm_psb(hipStream_t st, float* C, const float* A,
                            const unsigned* swzW, const float* bias,
                            const float* Res, int M, int N, int K, int flags)
{
    gemm(st, C, A, (const float*)swzW, bias, Res, M, N, K, K, N, N,
         1, 1, 0, 0, 0, 0, 0, 0, 0, 0, flags, 1);
}

static inline void swz(hipStream_t st, unsigned* dst, const float* W, int K, int N)
{
    swizzle_weight_kernel<<<dim3((K / 32) * (N / 128)), 256, 0, st>>>(dst, W, K, N);
}

static inline void lnorm(hipStream_t st, float* out, const float* in, LnP p,
                         int T, int F, int doGelu = 0)
{
    layernorm_kernel<<<dim3(T), 256, 0, st>>>(out, in, p.g, p.b, F, doGelu);
}

static void run_block(hipStream_t st, const Blk& p, float* x,
                      const float* keyIn, const float* valIn, int L,
                      Scratch& t, SwzW& w)
{
    const int T = BATCH * L;
    const int Z = BATCH * NHEAD;
    const long long LD = (long long)L * D_MODEL;
    const long long LL = (long long)L * L;

    // ---- pre-swizzle this block's weights (once, reused by every M-tile) ----
    swz(st, w.q, p.q.W, D_MODEL, D_MODEL);
    swz(st, w.k, p.k.W, D_MODEL, D_MODEL);
    swz(st, w.v, p.v.W, D_MODEL, D_MODEL);
    swz(st, w.proj, p.proj.W, D_MODEL, D_MODEL);
    swz(st, w.m0, p.mlp0.W, D_MODEL, 4 * D_MODEL);
    swz(st, w.m1, p.mlp1.W, 4 * D_MODEL, 4 * D_MODEL);
    swz(st, w.m2, p.mlp2.W, 4 * D_MODEL, D_MODEL);

    // ---- cross attention ----
    lnorm(st, t.ln, x, p.ln1_q, T, D_MODEL);
    gemm_psb(st, t.q, t.ln, w.q, p.q.b, nullptr, T, D_MODEL, D_MODEL, 0);
    lnorm(st, t.ln, keyIn, p.ln1_k, T, D_MODEL);
    gemm_psb(st, t.k, t.ln, w.k, p.k.b, nullptr, T, D_MODEL, D_MODEL, 0);
    lnorm(st, t.ln, valIn, p.ln1_v, T, D_MODEL);
    gemm_psb(st, t.v, t.ln, w.v, p.v.b, nullptr, T, D_MODEL, D_MODEL, 0);

    gemm(st, t.s, t.q, t.k, nullptr, nullptr, L, L, HDIM,
         D_MODEL, D_MODEL, L, Z, NHEAD,
         LD, HDIM, LD, HDIM, (long long)NHEAD * LL, LL, 0, /*transB=*/1, 0);
    attn_softmax_kernel<<<dim3(L, Z), 32, 0, st>>>(t.s, p.cross_bias, p.rel_bias, L, L, 0);
    gemm(st, t.y, t.s, t.v, nullptr, nullptr, L, HDIM, L,
         L, D_MODEL, D_MODEL, Z, NHEAD,
         (long long)NHEAD * LL, LL, LD, HDIM, LD, HDIM, 0, 0, 0);
    gemm_psb(st, t.o1, t.y, w.proj, p.proj.b, x, T, D_MODEL, D_MODEL, EP_RES);

    // ---- self attention ----
    lnorm(st, t.ln, t.o1, p.ln_self, T, D_MODEL);
    gemm_psb(st, t.q, t.ln, w.q, p.q.b, nullptr, T, D_MODEL, D_MODEL, 0);
    gemm_psb(st, t.k, t.ln, w.k, p.k.b, nullptr, T, D_MODEL, D_MODEL, 0);
    gemm_psb(st, t.v, t.ln, w.v, p.v.b, nullptr, T, D_MODEL, D_MODEL, 0);
    gemm(st, t.s, t.q, t.k, nullptr, nullptr, L, L, HDIM,
         D_MODEL, D_MODEL, L, Z, NHEAD,
         LD, HDIM, LD, HDIM, (long long)NHEAD * LL, LL, 0, 1, 0);
    attn_softmax_kernel<<<dim3(L, Z), 32, 0, st>>>(t.s, p.cross_bias, p.rel_bias, L, L, 1);
    gemm(st, t.y, t.s, t.v, nullptr, nullptr, L, HDIM, L,
         L, D_MODEL, D_MODEL, Z, NHEAD,
         (long long)NHEAD * LL, LL, LD, HDIM, LD, HDIM, 0, 0, 0);
    gemm_psb(st, t.o2, t.y, w.proj, p.proj.b, t.o1, T, D_MODEL, D_MODEL, EP_RES);

    // ---- MLP ----
    lnorm(st, t.ln, t.o2, p.ln2, T, D_MODEL);
    gemm_psb(st, t.h1, t.ln, w.m0, p.mlp0.b, nullptr, T, 4 * D_MODEL, D_MODEL, EP_GELU);
    gemm_psb(st, t.h2, t.h1, w.m1, p.mlp1.b, nullptr, T, 4 * D_MODEL, 4 * D_MODEL, EP_GELU);
    gemm_psb(st, x, t.h2, w.m2, p.mlp2.b, t.o2, T, D_MODEL, 4 * D_MODEL, EP_RES);
}

extern "C" void kernel_launch(void* const* d_in, const int* in_sizes, int n_in,
                              void* d_out, int out_size, void* d_ws, size_t ws_size,
                              hipStream_t stream)
{
    (void)in_sizes; (void)n_in; (void)out_size; (void)ws_size;

    // ---- walk inputs in setup_inputs() dict insertion order ----
    int cur = 0;
    auto nxt = [&]() -> const float* { return (const float*)d_in[cur++]; };
    const float* H_real = nxt();  const float* H_imag = nxt();
    const float* W_real = nxt();  const float* W_imag = nxt();
    const float* P_real = nxt();  const float* P_imag = nxt();
    auto lin = [&]() { Lin l; l.W = nxt(); l.b = nxt(); return l; };
    auto lnp = [&]() { LnP l; l.g = nxt(); l.b = nxt(); return l; };
    Lin ant_ch = lin(), ant_bm = lin(), ant_pr = lin();
    Lin user_ch = lin(), user_bm = lin(), user_pr = lin();
    auto blk = [&]() {
        Blk b;
        b.q = lin(); b.k = lin(); b.v = lin(); b.proj = lin();
        b.cross_bias = nxt(); b.rel_bias = nxt();
        b.ln_self = lnp(); b.ln1_q = lnp(); b.ln1_k = lnp(); b.ln1_v = lnp(); b.ln2 = lnp();
        b.mlp0 = lin(); b.mlp1 = lin(); b.mlp2 = lin();
        return b;
    };
    Blk ant[NLAYER], usr[NLAYER];
    for (int j = 0; j < NLAYER; ++j) ant[j] = blk();
    for (int j = 0; j < NLAYER; ++j) usr[j] = blk();
    Lin l1 = lin(); LnP oln1 = lnp(); Lin l2 = lin(); LnP oln2 = lnp(); Lin l3 = lin();

    // ---- carve workspace ----
    char* wp = (char*)d_ws;
    auto carveB = [&](size_t bytes) -> char* {
        char* p = wp;
        wp += (bytes + 255) & ~(size_t)255;
        return p;
    };
    auto carve = [&](size_t nf) -> float* { return (float*)carveB(nf * 4); };
    const int LA = 2 * NTX;                 // 128 antenna tokens
    const int LU = 2 * KUSR;                // 32  user tokens
    const size_t TA = (size_t)BATCH * LA;   // 8192 rows
    const size_t TU = (size_t)BATCH * LU;   // 2048 rows

    float* antH = carve(TA * KUSR); float* antW = carve(TA * KUSR); float* antP = carve(TA * KUSR);
    float* usrH = carve(TU * NTX);  float* usrW = carve(TU * NTX);  float* usrP = carve(TU * NTX);
    float* xA = carve(TA * D_MODEL); float* wA = carve(TA * D_MODEL); float* pA = carve(TA * D_MODEL);
    float* xU = carve(TU * D_MODEL); float* wU = carve(TU * D_MODEL); float* pU = carve(TU * D_MODEL);
    Scratch t;
    t.ln = carve(TA * D_MODEL); t.q = carve(TA * D_MODEL);
    t.k  = carve(TA * D_MODEL); t.v = carve(TA * D_MODEL);
    t.s  = carve((size_t)BATCH * NHEAD * LA * LA);
    t.y  = carve(TA * D_MODEL); t.o1 = carve(TA * D_MODEL); t.o2 = carve(TA * D_MODEL);
    t.h1 = carve(TA * 4 * D_MODEL); t.h2 = carve(TA * 4 * D_MODEL);
    float* h1  = carve((size_t)BATCH * 4 * D_MODEL);
    float* h1n = carve((size_t)BATCH * 4 * D_MODEL);
    float* h2  = carve((size_t)BATCH * 2 * D_MODEL);
    float* h2n = carve((size_t)BATCH * 2 * D_MODEL);
    float* vec = carve((size_t)BATCH * 2 * KUSR * NTX);
    // pre-swizzled bf16 weight buffers (reused across blocks; dwords = K*N/2)
    SwzW sw;
    sw.q    = (unsigned*)carveB((size_t)D_MODEL * D_MODEL * 2);
    sw.k    = (unsigned*)carveB((size_t)D_MODEL * D_MODEL * 2);
    sw.v    = (unsigned*)carveB((size_t)D_MODEL * D_MODEL * 2);
    sw.proj = (unsigned*)carveB((size_t)D_MODEL * D_MODEL * 2);
    sw.m0   = (unsigned*)carveB((size_t)D_MODEL * 4 * D_MODEL * 2);
    sw.m1   = (unsigned*)carveB((size_t)4 * D_MODEL * 4 * D_MODEL * 2);
    sw.m2   = (unsigned*)carveB((size_t)4 * D_MODEL * D_MODEL * 2);

    // ---- build stream inputs ----
    int totA = (int)(TA * KUSR), totU = (int)(TU * NTX);
    gather_stream_kernel<<<(totA + 255) / 256, 256, 0, stream>>>(antH, H_real, H_imag, 0, totA);
    gather_stream_kernel<<<(totA + 255) / 256, 256, 0, stream>>>(antW, W_real, W_imag, 0, totA);
    gather_stream_kernel<<<(totA + 255) / 256, 256, 0, stream>>>(antP, P_real, P_imag, 0, totA);
    gather_stream_kernel<<<(totU + 255) / 256, 256, 0, stream>>>(usrH, H_real, H_imag, 1, totU);
    gather_stream_kernel<<<(totU + 255) / 256, 256, 0, stream>>>(usrW, W_real, W_imag, 1, totU);
    gather_stream_kernel<<<(totU + 255) / 256, 256, 0, stream>>>(usrP, P_real, P_imag, 1, totU);

    // ---- embeddings (K padded to 32 inside GEMM; slow path) ----
    gemm_tok(stream, xA, antH, ant_ch.W, ant_ch.b, nullptr, (int)TA, D_MODEL, KUSR, 0);
    gemm_tok(stream, wA, antW, ant_bm.W, ant_bm.b, nullptr, (int)TA, D_MODEL, KUSR, 0);
    gemm_tok(stream, pA, antP, ant_pr.W, ant_pr.b, nullptr, (int)TA, D_MODEL, KUSR, 0);
    gemm_tok(stream, xU, usrH, user_ch.W, user_ch.b, nullptr, (int)TU, D_MODEL, NTX, 0);
    gemm_tok(stream, wU, usrW, user_bm.W, user_bm.b, nullptr, (int)TU, D_MODEL, NTX, 0);
    gemm_tok(stream, pU, usrP, user_pr.W, user_pr.b, nullptr, (int)TU, D_MODEL, NTX, 0);

    // ---- transformer stacks ----
    for (int j = 0; j < NLAYER; ++j) run_block(stream, ant[j], xA, wA, pA, LA, t, sw);
    for (int j = 0; j < NLAYER; ++j) run_block(stream, usr[j], xU, wU, pU, LU, t, sw);

    // ---- output head: concat handled as split-K over l1 weight rows ----
    // (l1 weights used once with M=64 -> direct fp32 path is the bandwidth optimum)
    const int KA = LA * D_MODEL;   // 65536
    const int KU = LU * D_MODEL;   // 16384
    gemm_tok(stream, h1, xA, l1.W, l1.b, nullptr, BATCH, 4 * D_MODEL, KA, 0);
    gemm_tok(stream, h1, xU, l1.W + (size_t)KA * 4 * D_MODEL, nullptr, nullptr,
             BATCH, 4 * D_MODEL, KU, EP_ACC);
    lnorm(stream, h1n, h1, oln1, BATCH, 4 * D_MODEL, /*gelu=*/1);
    gemm_tok(stream, h2, h1n, l2.W, l2.b, nullptr, BATCH, 2 * D_MODEL, 4 * D_MODEL, 0);
    lnorm(stream, h2n, h2, oln2, BATCH, 2 * D_MODEL, /*gelu=*/1);
    gemm_tok(stream, vec, h2n, l3.W, l3.b, nullptr, BATCH, 2 * KUSR * NTX, 2 * D_MODEL, 0);

    finalize_kernel<<<BATCH, 256, 0, stream>>>((float*)d_out, vec);
}
